// YakMoE_50079318672051
// MI455X (gfx1250) — matrix operand order repaired and checked by reading, not compiled, not run on hardware
//
#include <hip/hip_runtime.h>

// ---------------- problem constants ----------------
#define H_DIM 2048
#define I_DIM 4096
#define E_NUM 8
#define T_NUM 2048
#define TM    16              // token tile per workgroup
#define NW    16              // waves per workgroup
#define NTHREADS (NW * 32)

#if __has_builtin(__builtin_amdgcn_global_load_async_to_lds_b128)
#define HAVE_ASYNC_LDS 1
#else
#define HAVE_ASYNC_LDS 0
#endif

// ---------------- WMMA types ----------------
typedef __attribute__((ext_vector_type(16))) __bf16 v16bf;
typedef __attribute__((ext_vector_type(8)))  float  v8f;

// builtin async-copy payload type (per hipcc diagnostic: 'int __vector(4)')
typedef int v4i __attribute__((vector_size(16)));
typedef __attribute__((address_space(1))) v4i* gv4i_p;   // global (AS1)
typedef __attribute__((address_space(3))) v4i* lv4i_p;   // LDS (AS3)

union F16U { v16bf v; __bf16 b[16]; unsigned short u[16]; uint4 q[2]; };
union F8U  { v8f v; float f[8]; };

// native f32 -> bf16 (hardware cvt, round-to-nearest-even)
__device__ __forceinline__ unsigned short f2bf(float x) {
  __bf16 b = (__bf16)x;
  return __builtin_bit_cast(unsigned short, b);
}

// A fragment (16x32 bf16, M x K) from LDS row-major [16][ld].
// ISA 7.12.2: lane<16 -> row=lane, K in {0..7,16..23}; lane>=16 -> row=lane-16, K in {8..15,24..31}
__device__ __forceinline__ v16bf lds_fragA(const unsigned short* base, int ld, int k0, int lane) {
  const int r  = lane & 15;
  const int kb = (lane >> 4) << 3;                    // 0 or 8
  const unsigned short* p = base + r * ld + k0 + kb;
  F16U f;
  f.q[0] = *(const uint4*)(p);                        // K = k0+kb .. +7
  f.q[1] = *(const uint4*)(p + 16);                   // K = k0+kb+16 .. +23
  return f.v;
}

// B fragment (32x16 bf16, K x N) streamed from global fp32 weights stored [N][ldk] row-major.
// lane<16 -> col=lane, K in {k0..k0+7, k0+16..k0+23} ; lane>=16 -> col=lane-16, K in {k0+8..k0+15, k0+24..k0+31}
__device__ __forceinline__ v16bf glb_fragB(const float* __restrict__ base, int ldk, int k0, int lane) {
  const int n  = lane & 15;
  const int kb = (lane >> 4) << 3;                    // 0 or 8
  const float* p = base + (size_t)n * ldk + k0 + kb;
  __builtin_prefetch(p + 64, 0, 1);                   // global_prefetch_b8: pull next K chunk
  float4 a = *(const float4*)(p + 0);
  float4 b = *(const float4*)(p + 4);
  float4 c = *(const float4*)(p + 16);
  float4 d = *(const float4*)(p + 20);
  F16U f;
  f.b[0]=(__bf16)a.x;  f.b[1]=(__bf16)a.y;  f.b[2]=(__bf16)a.z;  f.b[3]=(__bf16)a.w;
  f.b[4]=(__bf16)b.x;  f.b[5]=(__bf16)b.y;  f.b[6]=(__bf16)b.z;  f.b[7]=(__bf16)b.w;
  f.b[8]=(__bf16)c.x;  f.b[9]=(__bf16)c.y;  f.b[10]=(__bf16)c.z; f.b[11]=(__bf16)c.w;
  f.b[12]=(__bf16)d.x; f.b[13]=(__bf16)d.y; f.b[14]=(__bf16)d.z; f.b[15]=(__bf16)d.w;
  return f.v;
}

// ---------------- router: logits -> softmax -> top2 -> renorm ----------------
__global__ void yak_router(const float* __restrict__ x, const float* __restrict__ gw,
                           int* __restrict__ tid_out, float* __restrict__ tw_out) {
  const int wave = threadIdx.x >> 5;
  const int lane = threadIdx.x & 31;
  const int t = blockIdx.x * (blockDim.x >> 5) + wave;
  if (t >= T_NUM) return;

  float acc[E_NUM];
#pragma unroll
  for (int e = 0; e < E_NUM; ++e) acc[e] = 0.f;
  const float* xt = x + (size_t)t * H_DIM;
  for (int h = lane; h < H_DIM; h += 32) {
    float xv = xt[h];
#pragma unroll
    for (int e = 0; e < E_NUM; ++e) acc[e] += xv * gw[e * H_DIM + h];
  }
#pragma unroll
  for (int e = 0; e < E_NUM; ++e) {
#pragma unroll
    for (int off = 16; off > 0; off >>= 1) acc[e] += __shfl_xor(acc[e], off, 32);
  }
  if (lane == 0) {
    float m = acc[0];
#pragma unroll
    for (int e = 1; e < E_NUM; ++e) m = fmaxf(m, acc[e]);
    float p[E_NUM]; float s = 0.f;
#pragma unroll
    for (int e = 0; e < E_NUM; ++e) { p[e] = __expf(acc[e] - m); s += p[e]; }
    float inv = __builtin_amdgcn_rcpf(s);
#pragma unroll
    for (int e = 0; e < E_NUM; ++e) p[e] *= inv;
    int b0 = 0;
#pragma unroll
    for (int e = 1; e < E_NUM; ++e) if (p[e] > p[b0]) b0 = e;
    int b1 = (b0 == 0) ? 1 : 0;
#pragma unroll
    for (int e = 0; e < E_NUM; ++e) if (e != b0 && p[e] > p[b1]) b1 = e;
    float w0 = p[b0], w1 = p[b1];
    float rn = __builtin_amdgcn_rcpf(w0 + w1);
    tid_out[t * 2 + 0] = b0;
    tid_out[t * 2 + 1] = b1;
    tw_out[t * 2 + 0] = w0 * rn;
    tw_out[t * 2 + 1] = w1 * rn;
  }
}

// ---------------- fused expert MLP ----------------
// One workgroup owns a 16-token tile. Loops experts; phase1 builds the scaled
// SwiGLU activation tile (16 x I) in LDS via bf16 WMMA, phase2 accumulates the
// down projection into fp32 register accumulators; output written once.
__global__ __launch_bounds__(NTHREADS)
void yak_moe(const float* __restrict__ x, const float* __restrict__ wgu,
             const float* __restrict__ w2, const int* __restrict__ tids,
             const float* __restrict__ tws, float* __restrict__ out) {
  const int LDX = H_DIM + 8;        // bf16 elements; keeps 16B alignment
  const int LDA = I_DIM + 8;
  __shared__ __align__(16) unsigned short xs[TM * (H_DIM + 8)];   // 64.3 KB
  __shared__ __align__(16) unsigned short as[TM * (I_DIM + 8)];   // 128.3 KB
  __shared__ float sc[TM];

  const int tid  = threadIdx.x;
  const int wave = tid >> 5;
  const int lane = tid & 31;
  const int t0   = blockIdx.x * TM;

#if HAVE_ASYNC_LDS
  // Stage the fp32 x tile into LDS with the CDNA5 async-DMA path (reusing the
  // `as` buffer, which is not live yet), then convert LDS fp32 -> LDS bf16.
  {
    float* asf = (float*)as;                    // 16*2048 fp32 = 128KB fits in `as`
    const float* src = x + (size_t)t0 * H_DIM;
    for (int idx = tid; idx < TM * H_DIM / 4; idx += NTHREADS) {
      __builtin_amdgcn_global_load_async_to_lds_b128(
          (gv4i_p)(src + idx * 4), (lv4i_p)(asf + idx * 4), 0, 0);
    }
    asm volatile("s_wait_asynccnt 0x0" ::: "memory");
    // each thread converts exactly the addresses it DMA'd (per-wave wait suffices)
    for (int idx = tid; idx < TM * H_DIM / 4; idx += NTHREADS) {
      float4 v = *(const float4*)(asf + idx * 4);
      int row = idx / (H_DIM / 4);
      int c4  = idx % (H_DIM / 4);
      ushort4 o;
      o.x = f2bf(v.x); o.y = f2bf(v.y); o.z = f2bf(v.z); o.w = f2bf(v.w);
      *(ushort4*)(xs + row * LDX + c4 * 4) = o;
    }
  }
#else
  for (int idx = tid; idx < TM * H_DIM / 4; idx += NTHREADS) {
    int row = idx / (H_DIM / 4);
    int c4  = idx % (H_DIM / 4);
    float4 v = *(const float4*)(x + (size_t)(t0 + row) * H_DIM + c4 * 4);
    ushort4 o;
    o.x = f2bf(v.x); o.y = f2bf(v.y); o.z = f2bf(v.z); o.w = f2bf(v.w);
    *(ushort4*)(xs + row * LDX + c4 * 4) = o;
  }
#endif

  // per-wave output accumulators: 8 tiles of 16x16 fp32 (C layout, 8 VGPR each)
  F8U acc[8];
#pragma unroll
  for (int j = 0; j < 8; ++j)
#pragma unroll
    for (int r = 0; r < 8; ++r) acc[j].f[r] = 0.f;

  const int mrow = (lane >> 4) << 3;   // C/D layout: row = mrow + vgpr, col = lane&15
  const int ncol = lane & 15;

  for (int e = 0; e < E_NUM; ++e) {
    __syncthreads();                   // protect `as` (prev phase2 / staging) and `sc`
    if (tid < TM) {
      int t = t0 + tid;
      float s = 0.f;
      if (tids[t * 2 + 0] == e) s += tws[t * 2 + 0];
      if (tids[t * 2 + 1] == e) s += tws[t * 2 + 1];
      sc[tid] = s;
    }
    __syncthreads();
    float tot = 0.f;
#pragma unroll
    for (int m = 0; m < TM; ++m) tot += sc[m];
    if (tot == 0.f) continue;          // uniform: nobody in this tile picked e

    // ---- phase 1: act[16][I] = silu(x@Wg^T) * (x@Wu^T) * combine ----
    const float* wg = wgu + (size_t)e * 2 * I_DIM * H_DIM;   // [2I][H] row-major
    for (int ft = wave; ft < I_DIM / 16; ft += NW) {
      const int i0 = ft * 16;
      const float* wgr = wg + (size_t)i0 * H_DIM;            // gate rows
      const float* wur = wg + (size_t)(I_DIM + i0) * H_DIM;  // up rows
      F8U dg, du;
#pragma unroll
      for (int r = 0; r < 8; ++r) { dg.f[r] = 0.f; du.f[r] = 0.f; }
      for (int k0 = 0; k0 < H_DIM; k0 += 32) {
        v16bf a  = lds_fragA(xs, LDX, k0, lane);
        v16bf bg = glb_fragB(wgr, H_DIM, k0, lane);
        v16bf bu = glb_fragB(wur, H_DIM, k0, lane);
        dg.v = __builtin_amdgcn_wmma_f32_16x16x32_bf16(false, a, false, bg, (short)0, dg.v, false, false);
        du.v = __builtin_amdgcn_wmma_f32_16x16x32_bf16(false, a, false, bu, (short)0, du.v, false, false);
      }
#pragma unroll
      for (int r = 0; r < 8; ++r) {
        int m = mrow + r;
        float g = dg.f[r];
        // silu via fast reciprocal (v_rcp_f32) instead of precise-division chain
        float sig = __builtin_amdgcn_rcpf(1.f + __expf(-g));
        float act = g * sig * du.f[r] * sc[m];
        as[m * LDA + i0 + ncol] = f2bf(act);
      }
    }
    __syncthreads();

    // ---- phase 2: out += act @ W2^T (accumulate in registers) ----
    const float* w2e = w2 + (size_t)e * H_DIM * I_DIM;       // [H][I] row-major
#pragma unroll
    for (int j = 0; j < 8; ++j) {
      const int h0 = (wave + j * NW) * 16;
      const float* w2r = w2e + (size_t)h0 * I_DIM;
      v8f c = acc[j].v;
      for (int k0 = 0; k0 < I_DIM; k0 += 32) {
        v16bf a = lds_fragA(as, LDA, k0, lane);
        v16bf b = glb_fragB(w2r, I_DIM, k0, lane);
        c = __builtin_amdgcn_wmma_f32_16x16x32_bf16(false, a, false, b, (short)0, c, false, false);
      }
      acc[j].v = c;
    }
  }

  // single write of the fully-combined output tile
#pragma unroll
  for (int j = 0; j < 8; ++j) {
    const int h0 = (wave + j * NW) * 16;
#pragma unroll
    for (int r = 0; r < 8; ++r) {
      out[(size_t)(t0 + mrow + r) * H_DIM + h0 + ncol] = acc[j].f[r];
    }
  }
}

// ---------------- launcher ----------------
extern "C" void kernel_launch(void* const* d_in, const int* in_sizes, int n_in,
                              void* d_out, int out_size, void* d_ws, size_t ws_size,
                              hipStream_t stream) {
  (void)in_sizes; (void)n_in; (void)out_size; (void)ws_size;
  const float* x   = (const float*)d_in[0];   // [T, H]
  const float* gw  = (const float*)d_in[1];   // [E, H]
  const float* wgu = (const float*)d_in[2];   // [E, 2I, H]
  const float* w2  = (const float*)d_in[3];   // [E, H, I]
  float* out = (float*)d_out;                 // [T, H]

  int*   tids = (int*)d_ws;                                   // [T,2]
  float* tws  = (float*)((char*)d_ws + T_NUM * 2 * sizeof(int));

  yak_router<<<T_NUM / 8, 256, 0, stream>>>(x, gw, tids, tws);
  yak_moe<<<T_NUM / TM, NTHREADS, 0, stream>>>(x, wgu, w2, tids, tws, out);
}
